// AgentAttention_1262720385546
// MI455X (gfx1250) — matrix operand order repaired and checked by reading, not compiled, not run on hardware
//
#include <hip/hip_runtime.h>
#include <hip/hip_bf16.h>

// ---------------------------------------------------------------------------
// Agent Attention for MI455X (gfx1250), wave32 + WMMA f32_16x16x32_f16.
// Large GEMMs use 2x2 register-blocked wave tiles (32x32 output per wave):
// each A/B fragment feeds two WMMAs -> 2 b128 loads per WMMA instead of 4.
// ---------------------------------------------------------------------------

typedef __attribute__((ext_vector_type(16))) _Float16 v16h;
typedef __attribute__((ext_vector_type(8)))  float    v8f;

#define NB 32
#define NH 6
#define HD 32
#define CC 192
#define HW 56
#define NN 3136       // 56*56
#define NA 49
#define AP 64         // padded agent dim
#define SCALE 0.17677669529663688f   // 1/sqrt(32)

// ---- WMMA helpers ----------------------------------------------------------
// A fragment (16x32, 16-bit, row-major in memory): per ISA 7.12.2,
// lanes 0-15 hold K 0-7 / 16-23, lanes 16-31 hold K 8-15 / 24-31.
__device__ __forceinline__ v16h ld_a(const _Float16* __restrict__ row, int k0, int half) {
  union { v16h v; uint4 u[2]; } f;
  const _Float16* p = row + k0 + half * 8;
  f.u[0] = *reinterpret_cast<const uint4*>(p);
  f.u[1] = *reinterpret_cast<const uint4*>(p + 16);
  return f.v;
}
// B fragment (32x16, 16-bit) loaded from B^T stored N-major (rows = N, cols = K):
// lanes 0-15 hold K 0-7 / 16-23, lanes 16-31 hold K 8-15 / 24-31 (same as A).
__device__ __forceinline__ v16h ld_b(const _Float16* __restrict__ rowT, int k0, int half) {
  union { v16h v; uint4 u[2]; } f;
  const _Float16* p = rowT + k0 + half * 8;
  f.u[0] = *reinterpret_cast<const uint4*>(p);
  f.u[1] = *reinterpret_cast<const uint4*>(p + 16);
  return f.v;
}
__device__ __forceinline__ v8f wmma32(v16h a, v16h b, v8f c) {
  return __builtin_amdgcn_wmma_f32_16x16x32_f16(false, a, false, b, (short)0, c, false, false);
}

// ---- conversion kernels ----------------------------------------------------
__global__ void cvt_x_k(const float* __restrict__ x, _Float16* __restrict__ xh) {
  int t = blockIdx.x * 256 + threadIdx.x;           // exact: NB*NN*CC / 256
  int cc = t % CC; int bn = t / CC;
  int b = bn / NN; int nn = bn % NN;
  xh[t] = (_Float16)x[((size_t)b * CC + cc) * NN + nn];
}

__global__ void cvt_wqkv_k(const float* __restrict__ W, _Float16* __restrict__ wT) {
  int t = blockIdx.x * 256 + threadIdx.x;           // 576*192
  int k = t % CC; int j = t / CC;
  wT[t] = (_Float16)W[(size_t)k * (3 * CC) + j];
}

__global__ void cvt_projw_k(const float* __restrict__ W, _Float16* __restrict__ wT) {
  int t = blockIdx.x * 256 + threadIdx.x;           // 192*192
  int k = t % CC; int j = t / CC;
  wT[t] = (_Float16)W[(size_t)k * CC + j];
}

// ---- bias precompute: bilinear 7x7 -> 56x56 (half-pixel, edge clamp) ------
__global__ void bias_k(const float* __restrict__ anb, const float* __restrict__ nab,
                       const float* __restrict__ ahb, const float* __restrict__ awb,
                       const float* __restrict__ hab, const float* __restrict__ wab,
                       float* __restrict__ posb, float* __restrict__ agentb) {
  int t = blockIdx.x * 256 + threadIdx.x;
  if (t >= NH * NA * NN) return;
  int nn = t % NN; int ha = t / NN;
  int a = ha % NA; int h = ha / NA;
  int y = nn / HW, x = nn % HW;
  float sy = (y + 0.5f) * 0.125f - 0.5f;
  float fy = floorf(sy); float wy = sy - fy;
  int y0 = (int)fy, y1 = (int)fy + 1;
  y0 = y0 < 0 ? 0 : (y0 > 6 ? 6 : y0);
  y1 = y1 < 0 ? 0 : (y1 > 6 ? 6 : y1);
  float sx = (x + 0.5f) * 0.125f - 0.5f;
  float fx = floorf(sx); float wx = sx - fx;
  int x0 = (int)fx, x1 = (int)fx + 1;
  x0 = x0 < 0 ? 0 : (x0 > 6 ? 6 : x0);
  x1 = x1 < 0 ? 0 : (x1 > 6 ? 6 : x1);

  const float* im1 = anb + ((size_t)h * NA + a) * 49;
  float v1 = (1.f - wy) * ((1.f - wx) * im1[y0 * 7 + x0] + wx * im1[y0 * 7 + x1])
           +        wy  * ((1.f - wx) * im1[y1 * 7 + x0] + wx * im1[y1 * 7 + x1]);
  posb[((size_t)h * NA + a) * NN + nn] =
      v1 + ahb[((size_t)h * NA + a) * HW + y] + awb[((size_t)h * NA + a) * HW + x];

  const float* im2 = nab + ((size_t)h * NA + a) * 49;
  float v2 = (1.f - wy) * ((1.f - wx) * im2[y0 * 7 + x0] + wx * im2[y0 * 7 + x1])
           +        wy  * ((1.f - wx) * im2[y1 * 7 + x0] + wx * im2[y1 * 7 + x1]);
  agentb[((size_t)h * NN + nn) * NA + a] =
      v2 + hab[((size_t)h * HW + y) * NA + a] + wab[((size_t)h * HW + x) * NA + a];
}

// ---- qkv GEMM: (B*N x 192) @ (192 x 576), 2x2 wave tile, scatter q/k/v ----
__global__ void qkv_gemm_k(const _Float16* __restrict__ xh, const _Float16* __restrict__ wT,
                           const float* __restrict__ bqkv,
                           _Float16* __restrict__ qh, _Float16* __restrict__ kh,
                           float* __restrict__ vf, _Float16* __restrict__ vT) {
  int wid = blockIdx.x * 8 + (threadIdx.x >> 5);
  int lane = threadIdx.x & 31, half = lane >> 4, lc = lane & 15;
  int nt2 = wid % 18, mt2 = wid / 18;                // 3136 x 18 super-tiles
  int m0 = mt2 * 32, n0 = nt2 * 32;
  const _Float16* aRow0 = xh + (size_t)(m0 + lc) * CC;
  const _Float16* aRow1 = aRow0 + (size_t)16 * CC;
  const _Float16* bRow0 = wT + (size_t)(n0 + lc) * CC;
  const _Float16* bRow1 = bRow0 + (size_t)16 * CC;
  v8f acc00 = {}, acc01 = {}, acc10 = {}, acc11 = {};
#pragma unroll
  for (int k0 = 0; k0 < CC; k0 += 32) {
    v16h a0 = ld_a(aRow0, k0, half), a1 = ld_a(aRow1, k0, half);
    v16h b0 = ld_b(bRow0, k0, half), b1 = ld_b(bRow1, k0, half);
    acc00 = wmma32(a0, b0, acc00);
    acc01 = wmma32(a0, b1, acc01);
    acc10 = wmma32(a1, b0, acc10);
    acc11 = wmma32(a1, b1, acc11);
  }
#pragma unroll
  for (int mi = 0; mi < 2; ++mi) {
#pragma unroll
    for (int ni = 0; ni < 2; ++ni) {
      const v8f& acc = mi ? (ni ? acc11 : acc10) : (ni ? acc01 : acc00);
#pragma unroll
      for (int vg = 0; vg < 8; ++vg) {
        int row = m0 + mi * 16 + half * 8 + vg;
        int b = row / NN, nn = row % NN;
        int j = n0 + ni * 16 + lc;
        float val = acc[vg] + bqkv[j];
        int sel = j / CC, cj = j % CC, head = cj >> 5, d = cj & 31;
        size_t idx = ((size_t)(b * NH + head) * NN + nn) * HD + d;
        if (sel == 0)      qh[idx] = (_Float16)val;
        else if (sel == 1) kh[idx] = (_Float16)val;
        else {
          vf[((size_t)b * NN + nn) * CC + cj] = val;
          vT[((size_t)(b * NH + head) * HD + d) * NN + nn] = (_Float16)val;
        }
      }
    }
  }
}

// ---- agent pooling: 8x8 block mean of q, fold in softmax scale, pad to 64 -
__global__ void pool_k(const _Float16* __restrict__ qh, _Float16* __restrict__ agent) {
  int t = blockIdx.x * 256 + threadIdx.x;           // NB*NH*AP*HD = 393216
  int d = t & 31; int a = (t >> 5) & 63; int bh = t >> 11;
  _Float16* dst = agent + ((size_t)bh * AP + a) * HD + d;
  if (a >= NA) { *dst = (_Float16)0.f; return; }
  int by = a / 7, bx = a % 7;
  const _Float16* base = qh + (size_t)bh * NN * HD + d;
  float s = 0.f;
  for (int dy = 0; dy < 8; ++dy)
    for (int dx = 0; dx < 8; ++dx)
      s += (float)base[(size_t)((by * 8 + dy) * HW + bx * 8 + dx) * HD];
  *dst = (_Float16)(s * (SCALE / 64.f));
}

// ---- score1: (64 x 32) @ (32 x N) + pos_bias, 2x2 wave tile ----------------
__global__ void score1_k(const _Float16* __restrict__ agent, const _Float16* __restrict__ kh,
                         const float* __restrict__ posb, float* __restrict__ scores) {
  int wid = blockIdx.x * 8 + (threadIdx.x >> 5);
  int lane = threadIdx.x & 31, half = lane >> 4, lc = lane & 15;
  int nt2 = wid % 98; int mt2 = (wid / 98) & 1; int bh = wid / 196;
  int h = bh % NH;
  int m0 = mt2 * 32, n0 = nt2 * 32;
  const _Float16* aRow0 = agent + ((size_t)bh * AP + m0 + lc) * HD;
  const _Float16* aRow1 = aRow0 + (size_t)16 * HD;
  const _Float16* bRow0 = kh + ((size_t)bh * NN + n0 + lc) * HD;
  const _Float16* bRow1 = bRow0 + (size_t)16 * HD;
  v16h a0 = ld_a(aRow0, 0, half), a1 = ld_a(aRow1, 0, half);
  v16h b0 = ld_b(bRow0, 0, half), b1 = ld_b(bRow1, 0, half);
  v8f z = {};
  v8f acc00 = wmma32(a0, b0, z), acc01 = wmma32(a0, b1, z);
  v8f acc10 = wmma32(a1, b0, z), acc11 = wmma32(a1, b1, z);
#pragma unroll
  for (int mi = 0; mi < 2; ++mi) {
#pragma unroll
    for (int ni = 0; ni < 2; ++ni) {
      const v8f& acc = mi ? (ni ? acc11 : acc10) : (ni ? acc01 : acc00);
#pragma unroll
      for (int vg = 0; vg < 8; ++vg) {
        int ga = m0 + mi * 16 + half * 8 + vg;
        int gn = n0 + ni * 16 + lc;
        float v = acc[vg];
        if (ga < NA) v += posb[((size_t)h * NA + ga) * NN + gn];
        scores[((size_t)bh * AP + ga) * NN + gn] = v;
      }
    }
  }
}

// ---- softmax over n (3136), fp32 -> f16 ------------------------------------
__global__ void softmax1_k(const float* __restrict__ scores, _Float16* __restrict__ attn) {
  int row = blockIdx.x;                              // (b*NH+h)*AP + a
  int a = row & 63;
  int tid = threadIdx.x;
  const float* src = scores + (size_t)row * NN;
  _Float16* dst = attn + (size_t)row * NN;
  if (a >= NA) {
    for (int i = tid; i < NN; i += 256) dst[i] = (_Float16)0.f;
    return;
  }
  __shared__ float red[256];
  float vals[13];
  float m = -1e30f; int c = 0;
  for (int i = tid; i < NN; i += 256) { float v = src[i]; vals[c++] = v; m = fmaxf(m, v); }
  red[tid] = m; __syncthreads();
  for (int s = 128; s > 0; s >>= 1) { if (tid < s) red[tid] = fmaxf(red[tid], red[tid + s]); __syncthreads(); }
  m = red[0]; __syncthreads();
  float sum = 0.f; c = 0;
  for (int i = tid; i < NN; i += 256) { float e = __expf(vals[c] - m); vals[c++] = e; sum += e; }
  red[tid] = sum; __syncthreads();
  for (int s = 128; s > 0; s >>= 1) { if (tid < s) red[tid] += red[tid + s]; __syncthreads(); }
  float inv = 1.f / red[0];
  c = 0;
  for (int i = tid; i < NN; i += 256) dst[i] = (_Float16)(vals[c++] * inv);
}

// ---- agent_v: (64 x N) @ (N x 32), K loop of 98 WMMAs ----------------------
__global__ void agentv_k(const _Float16* __restrict__ attn, const _Float16* __restrict__ vT,
                         float* __restrict__ avf) {
  int wid = blockIdx.x * 8 + (threadIdx.x >> 5);
  int lane = threadIdx.x & 31, half = lane >> 4, lc = lane & 15;
  int nt = wid & 1; int mt = (wid >> 1) & 3; int bh = wid >> 3;
  const _Float16* aRow = attn + ((size_t)bh * AP + mt * 16 + lc) * NN;
  const _Float16* bRow = vT + ((size_t)bh * HD + nt * 16 + lc) * NN;
  v8f acc = {};
  for (int k0 = 0; k0 < NN; k0 += 32)
    acc = wmma32(ld_a(aRow, k0, half), ld_b(bRow, k0, half), acc);
#pragma unroll
  for (int vg = 0; vg < 8; ++vg) {
    int ga = mt * 16 + half * 8 + vg;
    int gd = nt * 16 + lc;
    avf[((size_t)bh * AP + ga) * HD + gd] = acc[vg];
  }
}

__global__ void avcvt_k(const float* __restrict__ avf, _Float16* __restrict__ avT) {
  int t = blockIdx.x * 256 + threadIdx.x;           // NB*NH*HD*AP = 393216
  int a = t & 63; int d = (t >> 6) & 31; int bh = t >> 11;
  avT[((size_t)bh * HD + d) * AP + a] =
      (a < NA) ? (_Float16)avf[((size_t)bh * AP + a) * HD + d] : (_Float16)0.f;
}

// ---- score2: (N x 32) @ (32 x 64) + agent_bias, 2x2 wave tile -------------
__global__ void score2_k(const _Float16* __restrict__ qh, const _Float16* __restrict__ agent,
                         const float* __restrict__ agentb, float* __restrict__ scores) {
  int wid = blockIdx.x * 8 + (threadIdx.x >> 5);
  int lane = threadIdx.x & 31, half = lane >> 4, lc = lane & 15;
  int nt2 = wid & 1; int mt2 = (wid >> 1) % 98; int bh = wid / 196;
  int h = bh % NH;
  int m0 = mt2 * 32, n0 = nt2 * 32;
  const _Float16* aRow0 = qh + ((size_t)bh * NN + m0 + lc) * HD;
  const _Float16* aRow1 = aRow0 + (size_t)16 * HD;
  const _Float16* bRow0 = agent + ((size_t)bh * AP + n0 + lc) * HD;
  const _Float16* bRow1 = bRow0 + (size_t)16 * HD;
  v16h a0 = ld_a(aRow0, 0, half), a1 = ld_a(aRow1, 0, half);
  v16h b0 = ld_b(bRow0, 0, half), b1 = ld_b(bRow1, 0, half);
  v8f z = {};
  v8f acc00 = wmma32(a0, b0, z), acc01 = wmma32(a0, b1, z);
  v8f acc10 = wmma32(a1, b0, z), acc11 = wmma32(a1, b1, z);
#pragma unroll
  for (int mi = 0; mi < 2; ++mi) {
#pragma unroll
    for (int ni = 0; ni < 2; ++ni) {
      const v8f& acc = mi ? (ni ? acc11 : acc10) : (ni ? acc01 : acc00);
#pragma unroll
      for (int vg = 0; vg < 8; ++vg) {
        int gn = m0 + mi * 16 + half * 8 + vg;
        int ga = n0 + ni * 16 + lc;
        float v = acc[vg];
        if (ga < NA) v += agentb[((size_t)h * NN + gn) * NA + ga];
        scores[((size_t)bh * NN + gn) * AP + ga] = v;
      }
    }
  }
}

// ---- softmax over agent dim (49 of 64) ------------------------------------
__global__ void softmax2_k(const float* __restrict__ scores, _Float16* __restrict__ attn) {
  int row = blockIdx.x;                              // (b*NH+h)*NN + nn
  int tid = threadIdx.x;                             // 64 threads
  __shared__ float red[64];
  float v = (tid < NA) ? scores[(size_t)row * AP + tid] : -1e30f;
  red[tid] = v; __syncthreads();
  for (int s = 32; s > 0; s >>= 1) { if (tid < s) red[tid] = fmaxf(red[tid], red[tid + s]); __syncthreads(); }
  float m = red[0]; __syncthreads();
  float e = (tid < NA) ? __expf(v - m) : 0.f;
  red[tid] = e; __syncthreads();
  for (int s = 32; s > 0; s >>= 1) { if (tid < s) red[tid] += red[tid + s]; __syncthreads(); }
  float inv = 1.f / red[0];
  attn[(size_t)row * AP + tid] = (_Float16)(e * inv);
}

// ---- out: (N x 64) @ (64 x 32), scatter to (b,n,c) ------------------------
__global__ void out_gemm_k(const _Float16* __restrict__ attn, const _Float16* __restrict__ avT,
                           float* __restrict__ outpre) {
  int wid = blockIdx.x * 8 + (threadIdx.x >> 5);
  int lane = threadIdx.x & 31, half = lane >> 4, lc = lane & 15;
  int nt = wid & 1; int mt = (wid >> 1) % 196; int bh = wid / 392;
  int b = bh / NH, h = bh % NH;
  const _Float16* aRow = attn + ((size_t)bh * NN + mt * 16 + lc) * AP;
  const _Float16* bRow = avT + ((size_t)bh * HD + nt * 16 + lc) * AP;
  v8f acc = {};
#pragma unroll
  for (int k0 = 0; k0 < AP; k0 += 32)
    acc = wmma32(ld_a(aRow, k0, half), ld_b(bRow, k0, half), acc);
#pragma unroll
  for (int vg = 0; vg < 8; ++vg) {
    int gn = mt * 16 + half * 8 + vg;
    int gd = nt * 16 + lc;
    outpre[((size_t)b * NN + gn) * CC + h * HD + gd] = acc[vg];
  }
}

// ---- depthwise 3x3 conv on v + residual add, emit f16 for proj GEMM -------
__global__ void dwconv_k(const float* __restrict__ vf, const float* __restrict__ dwcw,
                         const float* __restrict__ dwcb, const float* __restrict__ outpre,
                         _Float16* __restrict__ outh) {
  int t = blockIdx.x * 256 + threadIdx.x;           // NB*NN*CC exact
  int cc = t % CC; int bn = t / CC;
  int b = bn / NN; int nn = bn % NN;
  int y = nn / HW, x = nn % HW;
  float s = dwcb[cc];
  for (int ky = 0; ky < 3; ++ky) {
    int yy = y + ky - 1;
    if (yy < 0 || yy >= HW) continue;
    for (int kx = 0; kx < 3; ++kx) {
      int xx = x + kx - 1;
      if (xx < 0 || xx >= HW) continue;
      s += vf[((size_t)b * NN + yy * HW + xx) * CC + cc] * dwcw[cc * 9 + ky * 3 + kx];
    }
  }
  outh[t] = (_Float16)(outpre[t] + s);
}

// ---- proj GEMM: (B*N x 192) @ (192 x 192) + bias, 2x2 wave tile -----------
__global__ void proj_gemm_k(const _Float16* __restrict__ outh, const _Float16* __restrict__ wT,
                            const float* __restrict__ projb, float* __restrict__ dout) {
  int wid = blockIdx.x * 8 + (threadIdx.x >> 5);
  int lane = threadIdx.x & 31, half = lane >> 4, lc = lane & 15;
  int nt2 = wid % 6, mt2 = wid / 6;                  // 3136 x 6 super-tiles
  int m0 = mt2 * 32, n0 = nt2 * 32;
  const _Float16* aRow0 = outh + (size_t)(m0 + lc) * CC;
  const _Float16* aRow1 = aRow0 + (size_t)16 * CC;
  const _Float16* bRow0 = wT + (size_t)(n0 + lc) * CC;
  const _Float16* bRow1 = bRow0 + (size_t)16 * CC;
  v8f acc00 = {}, acc01 = {}, acc10 = {}, acc11 = {};
#pragma unroll
  for (int k0 = 0; k0 < CC; k0 += 32) {
    v16h a0 = ld_a(aRow0, k0, half), a1 = ld_a(aRow1, k0, half);
    v16h b0 = ld_b(bRow0, k0, half), b1 = ld_b(bRow1, k0, half);
    acc00 = wmma32(a0, b0, acc00);
    acc01 = wmma32(a0, b1, acc01);
    acc10 = wmma32(a1, b0, acc10);
    acc11 = wmma32(a1, b1, acc11);
  }
#pragma unroll
  for (int mi = 0; mi < 2; ++mi) {
#pragma unroll
    for (int ni = 0; ni < 2; ++ni) {
      const v8f& acc = mi ? (ni ? acc11 : acc10) : (ni ? acc01 : acc00);
#pragma unroll
      for (int vg = 0; vg < 8; ++vg) {
        int row = m0 + mi * 16 + half * 8 + vg;
        int b = row / NN, nn = row % NN;
        int cc = n0 + ni * 16 + lc;
        dout[((size_t)b * CC + cc) * NN + nn] = acc[vg] + projb[cc];
      }
    }
  }
}

// ---------------------------------------------------------------------------
extern "C" void kernel_launch(void* const* d_in, const int* in_sizes, int n_in,
                              void* d_out, int out_size, void* d_ws, size_t ws_size,
                              hipStream_t stream) {
  const float* x      = (const float*)d_in[0];
  const float* Wqkv   = (const float*)d_in[1];
  const float* bqkv   = (const float*)d_in[2];
  const float* proj_w = (const float*)d_in[3];
  const float* proj_b = (const float*)d_in[4];
  const float* dwc_w  = (const float*)d_in[5];
  const float* dwc_b  = (const float*)d_in[6];
  const float* an_b   = (const float*)d_in[7];
  const float* na_b   = (const float*)d_in[8];
  const float* ah_b   = (const float*)d_in[9];
  const float* aw_b   = (const float*)d_in[10];
  const float* ha_b   = (const float*)d_in[11];
  const float* wa_b   = (const float*)d_in[12];
  float* out = (float*)d_out;

  char* ws = (char*)d_ws;
  size_t off = 0;
  auto take = [&](size_t bytes) { char* p = ws + off; off += bytes; return p; };

  const size_t S_H16 = (size_t)NB * NN * CC * 2;      // 38.5 MB (f16 tensors)
  _Float16* xh     = (_Float16*)take(S_H16);          // x in (b,n,c); reused as outh
  _Float16* qh     = (_Float16*)take(S_H16);          // (b,h,n,d)
  _Float16* kh     = (_Float16*)take(S_H16);          // (b,h,n,d)
  _Float16* vT     = (_Float16*)take(S_H16);          // (b,h,d,n)
  float*    vf     = (float*)take((size_t)NB * NN * CC * 4);          // 77 MB
  float*    scores = (float*)take((size_t)NB * NH * AP * NN * 4);     // 154 MB (reused as outpre)
  _Float16* attn   = (_Float16*)take((size_t)NB * NH * AP * NN * 2);  // 77 MB (attn1 then attn2)
  _Float16* agent  = (_Float16*)take((size_t)NB * NH * AP * HD * 2);
  float*    avf    = (float*)take((size_t)NB * NH * AP * HD * 4);
  _Float16* avT    = (_Float16*)take((size_t)NB * NH * HD * AP * 2);
  float*    posb   = (float*)take((size_t)NH * NA * NN * 4);
  float*    agentb = (float*)take((size_t)NH * NN * NA * 4);
  _Float16* wqkvT  = (_Float16*)take((size_t)3 * CC * CC * 2);
  _Float16* projT  = (_Float16*)take((size_t)CC * CC * 2);
  _Float16* outh   = xh;            // reuse after qkv GEMM consumes xh
  float*    outpre = scores;        // reuse after softmax2 consumes scores

  // 1. conversions + bias precompute
  cvt_x_k<<<(NB * NN * CC) / 256, 256, 0, stream>>>(x, xh);
  cvt_wqkv_k<<<(3 * CC * CC) / 256, 256, 0, stream>>>(Wqkv, wqkvT);
  cvt_projw_k<<<(CC * CC) / 256, 256, 0, stream>>>(proj_w, projT);
  bias_k<<<(NH * NA * NN + 255) / 256, 256, 0, stream>>>(an_b, na_b, ah_b, aw_b, ha_b, wa_b,
                                                         posb, agentb);
  // 2. qkv GEMM (3136*18 super-tiles of 32x32, 8 waves/block)
  qkv_gemm_k<<<(3136 * 18) / 8, 256, 0, stream>>>(xh, wqkvT, bqkv, qh, kh, vf, vT);
  // 3. agent pooling (scale folded)
  pool_k<<<(NB * NH * AP * HD) / 256, 256, 0, stream>>>(qh, agent);
  // 4. agent_attn scores + softmax over n (2x2 wave tiles)
  score1_k<<<(NB * NH * 2 * 98) / 8, 256, 0, stream>>>(agent, kh, posb, scores);
  softmax1_k<<<NB * NH * AP, 256, 0, stream>>>(scores, attn);
  // 5. agent_v GEMM + transpose to f16
  agentv_k<<<(NB * NH * 8) / 8, 256, 0, stream>>>(attn, vT, avf);
  avcvt_k<<<(NB * NH * HD * AP) / 256, 256, 0, stream>>>(avf, avT);
  // 6. q_attn scores + softmax over agents (reuses scores + attn buffers)
  score2_k<<<(NB * NH * 98 * 2) / 8, 256, 0, stream>>>(qh, agent, agentb, scores);
  softmax2_k<<<NB * NH * NN, 64, 0, stream>>>(scores, attn);
  // 7. out = q_attn @ agent_v (writes fp32 outpre = scores buffer)
  out_gemm_k<<<(NB * NH * 196 * 2) / 8, 256, 0, stream>>>(attn, avT, outpre);
  // 8. depthwise conv residual, emit f16 (reuses xh buffer)
  dwconv_k<<<(NB * NN * CC) / 256, 256, 0, stream>>>(vf, dwc_w, dwc_b, outpre, outh);
  // 9. projection GEMM, scatter to (b,c,h,w)
  proj_gemm_k<<<(3136 * 6) / 8, 256, 0, stream>>>(outh, projT, proj_b, out);

  (void)in_sizes; (void)n_in; (void)out_size; (void)ws_size;
}
